// GPT3_33062658245209
// MI455X (gfx1250) — matrix operand order repaired and checked by reading, not compiled, not run on hardware
//
#include <hip/hip_runtime.h>

typedef __bf16 bf16;
typedef __attribute__((ext_vector_type(16))) __bf16 v16bf;
typedef __attribute__((ext_vector_type(8)))  float  v8f;

#define VOCAB  50257
#define SEQ    512
#define BATCH  2
#define NTOK   (BATCH * SEQ)
#define DMODEL 768
#define NLAYER 12
#define NHEAD  12
#define DHEAD  64
#define DFF    3072
#define EPSLN  1e-5f

// ---------------------------------------------------------------------------
// gfx1250 async global->LDS copy (32B per lane as 2x b128), ASYNCcnt-tracked.
// IOFFSET applies to both LDS dest and global src (ISA 08_async_tensor §4.4).
// Generic-pointer low 32 bits == LDS byte offset (aperture layout).
// ---------------------------------------------------------------------------
__device__ __forceinline__ void async_copy32(void* lds_dst, const void* gsrc) {
  unsigned lds_off = (unsigned)(unsigned long long)lds_dst;
  unsigned long long ga = (unsigned long long)gsrc;
  asm volatile(
      "global_load_async_to_lds_b128 %0, %1, off\n\t"
      "global_load_async_to_lds_b128 %0, %1, off offset:16"
      :
      : "v"(lds_off), "v"(ga)
      : "memory");
}

__device__ __forceinline__ void wait_async0() {
#if __has_builtin(__builtin_amdgcn_s_wait_asynccnt)
  __builtin_amdgcn_s_wait_asynccnt(0);
#else
  asm volatile("s_wait_asynccnt 0x0" ::: "memory");
#endif
}

// ---------------------------------------------------------------------------
// Embedding: x[t, :] = tok_emb[ids[t], :] + pos_emb[t % SEQ, :]
// ---------------------------------------------------------------------------
__global__ __launch_bounds__(256) void embed_kernel(
    const int* __restrict__ ids, const float* __restrict__ tok,
    const float* __restrict__ pos, float* __restrict__ x) {
  long long idx = (long long)blockIdx.x * blockDim.x + threadIdx.x;
  if (idx >= (long long)NTOK * DMODEL) return;
  int t = (int)(idx / DMODEL);
  int d = (int)(idx % DMODEL);
  int s = t % SEQ;
  x[idx] = tok[(long long)ids[t] * DMODEL + d] + pos[(long long)s * DMODEL + d];
}

// ---------------------------------------------------------------------------
// LayerNorm over D=768, one block (256 thr) per token; bf16 output for WMMA A.
// ---------------------------------------------------------------------------
__global__ __launch_bounds__(256) void ln_kernel(
    const float* __restrict__ x, const float* __restrict__ g,
    const float* __restrict__ b, bf16* __restrict__ out) {
  int t = blockIdx.x;
  const float* xr = x + (long long)t * DMODEL;
  __shared__ float red0[8];
  __shared__ float red1[8];
  int tid = threadIdx.x, lane = tid & 31, w = tid >> 5;
  float s = 0.f, ss = 0.f;
  for (int i = tid; i < DMODEL; i += 256) {
    float v = xr[i];
    s += v;
    ss += v * v;
  }
  for (int off = 16; off; off >>= 1) {
    s  += __shfl_xor(s, off, 32);
    ss += __shfl_xor(ss, off, 32);
  }
  if (lane == 0) { red0[w] = s; red1[w] = ss; }
  __syncthreads();
  float ts = 0.f, tss = 0.f;
#pragma unroll
  for (int i = 0; i < 8; ++i) { ts += red0[i]; tss += red1[i]; }
  float m   = ts / (float)DMODEL;
  float var = tss / (float)DMODEL - m * m;
  float inv = rsqrtf(var + EPSLN);
  bf16* orow = out + (long long)t * DMODEL;
  for (int i = tid; i < DMODEL; i += 256)
    orow[i] = (bf16)((xr[i] - m) * inv * g[i] + b[i]);
}

// ---------------------------------------------------------------------------
// Row-wise causal softmax: one wave per row of [B*H*S, SEQ]; bf16 probs out.
// ---------------------------------------------------------------------------
__global__ __launch_bounds__(256) void softmax_causal(
    const float* __restrict__ scores, bf16* __restrict__ probs) {
  int ridx = blockIdx.x * 8 + (threadIdx.x >> 5);
  if (ridx >= BATCH * NHEAD * SEQ) return;
  int i = ridx % SEQ;  // query position (causal limit)
  const float* sr = scores + (long long)ridx * SEQ;
  bf16* pr = probs + (long long)ridx * SEQ;
  int lane = threadIdx.x & 31;
  float vals[16];
  float mx = -3.0e38f;
#pragma unroll
  for (int j = 0; j < 16; ++j) {
    int c = j * 32 + lane;
    float v = (c <= i) ? sr[c] : -3.0e38f;
    vals[j] = v;
    mx = fmaxf(mx, v);
  }
  for (int off = 16; off; off >>= 1) mx = fmaxf(mx, __shfl_xor(mx, off, 32));
  float sum = 0.f;
#pragma unroll
  for (int j = 0; j < 16; ++j) {
    int c = j * 32 + lane;
    float e = (c <= i) ? expf(vals[j] - mx) : 0.f;
    vals[j] = e;
    sum += e;
  }
  for (int off = 16; off; off >>= 1) sum += __shfl_xor(sum, off, 32);
  float r = 1.f / sum;
#pragma unroll
  for (int j = 0; j < 16; ++j) pr[j * 32 + lane] = (bf16)(vals[j] * r);
}

// ---------------------------------------------------------------------------
// Double-buffered, LDS-tiled, register-blocked WMMA GEMM.
//   C[M,N] = act(scale * A @ B + bias) (+ residual)
//   Block tile 128x128, K-step 32, 2 LDS buffers. 256 threads = 8 waves
//   (2 M x 4 N); each wave computes a 64x32 sub-tile = 4x2 accumulators of
//   16x16 via v_wmma_f32_16x16x32_bf16 (8 WMMA per wave per K-step).
//   Pipeline: stage tile s+1 via global_load_async_to_lds_b128 (ASYNCcnt),
//   compute tile s from LDS, then s_wait_asynccnt 0 + one barrier per step —
//   the async fetch hides HBM latency under the matrix pipe.
//   B converted to bf16 into LDS transposed as Bs[n][k] (one column per
//   thread: single N-guard, contiguous 32B LDS store, lane-coalesced reads).
//   Fragment layouts (contiguous k, not ISA table order): A lanes 0-15 hold K0-15, lanes
//   16-31 hold K16-31; B lanes 0-15 hold K0-15, lanes 16-31 K16-31.
//   M must be a multiple of 128 (true for all call sites); N is guarded.
//   (b,h) batching via gridDim.y: off = (bh/H)*sXb + (bh%H)*sXh.
// ---------------------------------------------------------------------------
template <typename TB, bool TRANSB, bool GELU_ACT, bool OUT_BF16>
__global__ __launch_bounds__(256) void wmma_gemm(
    const bf16* __restrict__ A, const TB* __restrict__ B, void* __restrict__ C,
    const float* __restrict__ bias, const float* __restrict__ resid,
    int M, int N, int K, int lda, int ldb, int ldc, float scale, int H,
    long long sAb, long long sAh, long long sBb, long long sBh, long long sCb,
    long long sCh) {
  __shared__ bf16 As[2][128][32];  // [buf][m][k]
  __shared__ bf16 Bs[2][128][32];  // [buf][n][k] (transposed)

  int bh = blockIdx.y;
  const bf16* Ab = A + (long long)(bh / H) * sAb + (long long)(bh % H) * sAh;
  const TB* Bb   = B + (long long)(bh / H) * sBb + (long long)(bh % H) * sBh;
  long long co   = (long long)(bh / H) * sCb + (long long)(bh % H) * sCh;

  int tilesN = (N + 127) >> 7;
  int bm = (int)blockIdx.x / tilesN;
  int bn = (int)blockIdx.x % tilesN;
  int m0 = bm << 7;
  int n0 = bn << 7;

  int t = (int)threadIdx.x;
  int wave = t >> 5;
  int wm = wave >> 2;  // 0..1 : M half of block tile
  int wn = wave & 3;   // 0..3 : N quarter of block tile
  int lane = t & 31;
  int lh = lane >> 4;  // lane half
  int lr = lane & 15;

  // Staging assignment: 2 threads per row/column, 16 elements each.
  int sr   = t >> 1;        // A row within tile / B column within tile
  int sk   = (t & 1) << 4;  // k offset (0 or 16)
  int bcol = n0 + sr;       // global B column for this thread

  // Stage one 128x32 K-slab (A and B) into LDS buffer `buf`.
  auto stage = [&](int buf, int kb) {
    async_copy32(&As[buf][sr][sk], Ab + (long long)(m0 + sr) * lda + kb + sk);
    if constexpr (TRANSB && sizeof(TB) == 2) {
      // bf16 [N,K] rows: contiguous 32B, no conversion -> async copy.
      // (Used only when N % 128 == 0, so no guard needed.)
      async_copy32(&Bs[buf][sr][sk], Bb + (long long)bcol * ldb + kb + sk);
    } else if constexpr (TRANSB) {
      // f32 [N,K] rows: contiguous 64B, convert to bf16.
      v16bf tmp;
      if (bcol < N) {
        const float4* bg =
            (const float4*)(Bb + (long long)bcol * ldb + kb + sk);
#pragma unroll
        for (int q = 0; q < 4; ++q) {
          float4 f = bg[q];
          tmp[q * 4 + 0] = (bf16)f.x;
          tmp[q * 4 + 1] = (bf16)f.y;
          tmp[q * 4 + 2] = (bf16)f.z;
          tmp[q * 4 + 3] = (bf16)f.w;
        }
      } else {
#pragma unroll
        for (int e = 0; e < 16; ++e) tmp[e] = (bf16)0.0f;
      }
      *(v16bf*)&Bs[buf][sr][sk] = tmp;
    } else {
      // [K,N]: one column per thread (single guard), reads lane-coalesced
      // within each k-row, contiguous 32B LDS store.
      v16bf tmp;
      if (bcol < N) {
        const TB* bg = Bb + (long long)(kb + sk) * ldb + bcol;
#pragma unroll
        for (int e = 0; e < 16; ++e)
          tmp[e] = (bf16)(float)bg[(long long)e * ldb];
      } else {
#pragma unroll
        for (int e = 0; e < 16; ++e) tmp[e] = (bf16)0.0f;
      }
      *(v16bf*)&Bs[buf][sr][sk] = tmp;
    }
  };

  v8f acc[4][2] = {};

  // Prologue: stage tile 0 into buffer 0.
  stage(0, 0);
  wait_async0();
  __syncthreads();

  int nsteps = K >> 5;
  for (int s = 0; s < nsteps; ++s) {
    int cur = s & 1;
    bool more = (s + 1 < nsteps);
    // Issue async staging of the next slab; overlaps with compute below.
    if (more) stage(cur ^ 1, (s + 1) << 5);

    // ---- load fragments from LDS (buffer cur) ----
    v16bf af[4];
#pragma unroll
    for (int tm = 0; tm < 4; ++tm) {
      const bf16* ap = &As[cur][wm * 64 + tm * 16 + lr][lh * 16];
#pragma unroll
      for (int e = 0; e < 16; ++e) af[tm][e] = ap[e];
    }
    v16bf bf[2];
#pragma unroll
    for (int tn = 0; tn < 2; ++tn) {
      const bf16* bp = &Bs[cur][wn * 32 + tn * 16 + lr][lh * 16];
#pragma unroll
      for (int e = 0; e < 16; ++e) bf[tn][e] = bp[e];
    }
    // ---- 8 WMMAs ----
#pragma unroll
    for (int tm = 0; tm < 4; ++tm)
#pragma unroll
      for (int tn = 0; tn < 2; ++tn)
        acc[tm][tn] = __builtin_amdgcn_wmma_f32_16x16x32_bf16(
            false, af[tm], false, bf[tn], (short)0, acc[tm][tn], false, false);

    if (more) {
      wait_async0();    // next slab fully in LDS (this wave's async ops)
      __syncthreads();  // publish next slab; all waves done reading cur
    }
  }

  // ---- epilogue ----
  float* Cf = (float*)C;
  bf16* Ch = (bf16*)C;
#pragma unroll
  for (int tn = 0; tn < 2; ++tn) {
    int col = n0 + wn * 32 + tn * 16 + lr;
    if (col >= N) continue;
    float bvv = bias ? bias[col] : 0.f;
#pragma unroll
    for (int tm = 0; tm < 4; ++tm) {
#pragma unroll
      for (int r = 0; r < 8; ++r) {
        int orow = m0 + wm * 64 + tm * 16 + r + 8 * lh;
        float v = acc[tm][tn][r] * scale + bvv;
        if constexpr (GELU_ACT)
          v = 0.5f * v * (1.f + erff(v * 0.70710678118654752f));
        long long idx = co + (long long)orow * ldc + col;
        if (resid) v += resid[idx];
        if constexpr (OUT_BF16)
          Ch[idx] = (bf16)v;
        else
          Cf[idx] = v;
      }
    }
  }
}

// ---------------------------------------------------------------------------
// Host-side launch helper (M must be a multiple of 128)
// ---------------------------------------------------------------------------
template <typename TB, bool TRANSB, bool GELU_ACT, bool OUT_BF16>
static void gemm(hipStream_t st, const bf16* A, const TB* B, void* C,
                 const float* bias, const float* resid, int M, int N, int K,
                 int lda, int ldb, int ldc, float scale, int nbatch = 1,
                 int H = 1, long long sAb = 0, long long sAh = 0,
                 long long sBb = 0, long long sBh = 0, long long sCb = 0,
                 long long sCh = 0) {
  int tilesM = M >> 7;
  int tilesN = (N + 127) >> 7;
  dim3 grid(tilesM * tilesN, nbatch);
  wmma_gemm<TB, TRANSB, GELU_ACT, OUT_BF16><<<grid, 256, 0, st>>>(
      A, B, C, bias, resid, M, N, K, lda, ldb, ldc, scale, H, sAb, sAh, sBb,
      sBh, sCb, sCh);
}

extern "C" void kernel_launch(void* const* d_in, const int* in_sizes, int n_in,
                              void* d_out, int out_size, void* d_ws,
                              size_t ws_size, hipStream_t stream) {
  const int*   ids     = (const int*)d_in[0];
  const float* tok_emb = (const float*)d_in[1];
  const float* pos_emb = (const float*)d_in[2];
  const float* wq = (const float*)d_in[3];
  const float* bq = (const float*)d_in[4];
  const float* wk = (const float*)d_in[5];
  const float* bk = (const float*)d_in[6];
  const float* wv = (const float*)d_in[7];
  const float* bv = (const float*)d_in[8];
  const float* wo = (const float*)d_in[9];
  const float* bo = (const float*)d_in[10];
  const float* w1 = (const float*)d_in[11];
  const float* b1 = (const float*)d_in[12];
  const float* w2 = (const float*)d_in[13];
  const float* b2 = (const float*)d_in[14];
  const float* ln1_g = (const float*)d_in[15];
  const float* ln1_b = (const float*)d_in[16];
  const float* ln2_g = (const float*)d_in[17];
  const float* ln2_b = (const float*)d_in[18];
  const float* lnf_g = (const float*)d_in[19];
  const float* lnf_b = (const float*)d_in[20];

  // Workspace bump allocator (256B aligned slabs).
  char* wp = (char*)d_ws;
  auto alloc = [&](size_t bytes) -> void* {
    void* r = (void*)wp;
    wp += (bytes + 255) & ~(size_t)255;
    return r;
  };
  float* x  = (float*)alloc((size_t)NTOK * DMODEL * 4);          // residual
  bf16* hb  = (bf16*)alloc((size_t)NTOK * DMODEL * 2);           // LN out
  bf16* qb  = (bf16*)alloc((size_t)NTOK * DMODEL * 2);
  bf16* kb_ = (bf16*)alloc((size_t)NTOK * DMODEL * 2);
  bf16* vb  = (bf16*)alloc((size_t)NTOK * DMODEL * 2);
  bf16* ob  = (bf16*)alloc((size_t)NTOK * DMODEL * 2);
  float* sc = (float*)alloc((size_t)BATCH * NHEAD * SEQ * SEQ * 4);
  bf16* pb  = (bf16*)alloc((size_t)BATCH * NHEAD * SEQ * SEQ * 2);
  bf16* f1  = (bf16*)alloc((size_t)NTOK * DFF * 2);

  const long long sTD = (long long)SEQ * DMODEL;   // per-batch act stride
  const long long sSS = (long long)SEQ * SEQ;      // per-head score stride

  embed_kernel<<<((long long)NTOK * DMODEL + 255) / 256, 256, 0, stream>>>(
      ids, tok_emb, pos_emb, x);

  for (int l = 0; l < NLAYER; ++l) {
    const float* wq_ = wq + (long long)l * DMODEL * DMODEL;
    const float* wk_ = wk + (long long)l * DMODEL * DMODEL;
    const float* wv_ = wv + (long long)l * DMODEL * DMODEL;
    const float* wo_ = wo + (long long)l * DMODEL * DMODEL;
    const float* w1_ = w1 + (long long)l * DMODEL * DFF;
    const float* w2_ = w2 + (long long)l * DFF * DMODEL;
    const float* bq_ = bq + (long long)l * DMODEL;
    const float* bk_ = bk + (long long)l * DMODEL;
    const float* bv_ = bv + (long long)l * DMODEL;
    const float* bo_ = bo + (long long)l * DMODEL;
    const float* b1_ = b1 + (long long)l * DFF;
    const float* b2_ = b2 + (long long)l * DMODEL;

    // LN1 -> hb (bf16)
    ln_kernel<<<NTOK, 256, 0, stream>>>(x, ln1_g + (long long)l * DMODEL,
                                        ln1_b + (long long)l * DMODEL, hb);
    // Q, K, V projections (bf16 out, [B,S,D] layout)
    gemm<float, false, false, true>(stream, hb, wq_, qb, bq_, nullptr, NTOK,
                                    DMODEL, DMODEL, DMODEL, DMODEL, DMODEL,
                                    1.f);
    gemm<float, false, false, true>(stream, hb, wk_, kb_, bk_, nullptr, NTOK,
                                    DMODEL, DMODEL, DMODEL, DMODEL, DMODEL,
                                    1.f);
    gemm<float, false, false, true>(stream, hb, wv_, vb, bv_, nullptr, NTOK,
                                    DMODEL, DMODEL, DMODEL, DMODEL, DMODEL,
                                    1.f);
    // scores[b,h] = (Q_bh @ K_bh^T) / sqrt(dk); batched over 24 (b,h) pairs
    gemm<bf16, true, false, false>(
        stream, qb, kb_, sc, nullptr, nullptr, SEQ, SEQ, DHEAD, DMODEL, DMODEL,
        SEQ, 0.125f, BATCH * NHEAD, NHEAD, sTD, DHEAD, sTD, DHEAD,
        (long long)NHEAD * sSS, sSS);
    // causal softmax -> bf16 probs
    softmax_causal<<<(BATCH * NHEAD * SEQ + 7) / 8, 256, 0, stream>>>(sc, pb);
    // O[b,h] = P_bh @ V_bh  (bf16 out, scattered back into [B,S,D])
    gemm<bf16, false, false, true>(
        stream, pb, vb, ob, nullptr, nullptr, SEQ, DHEAD, SEQ, SEQ, DMODEL,
        DMODEL, 1.f, BATCH * NHEAD, NHEAD, (long long)NHEAD * sSS, sSS, sTD,
        DHEAD, sTD, DHEAD);
    // x += O @ Wo + bo
    gemm<float, false, false, false>(stream, ob, wo_, x, bo_, x, NTOK, DMODEL,
                                     DMODEL, DMODEL, DMODEL, DMODEL, 1.f);
    // LN2 -> hb
    ln_kernel<<<NTOK, 256, 0, stream>>>(x, ln2_g + (long long)l * DMODEL,
                                        ln2_b + (long long)l * DMODEL, hb);
    // f1 = gelu(hb @ W1 + b1) (bf16 out)
    gemm<float, false, true, true>(stream, hb, w1_, f1, b1_, nullptr, NTOK,
                                   DFF, DMODEL, DMODEL, DFF, DFF, 1.f);
    // x += f1 @ W2 + b2
    gemm<float, false, false, false>(stream, f1, w2_, x, b2_, x, NTOK, DMODEL,
                                     DFF, DFF, DMODEL, DMODEL, 1.f);
  }

  // Final LN, then tied-embedding logits: [NTOK, VOCAB] f32 into d_out.
  ln_kernel<<<NTOK, 256, 0, stream>>>(x, lnf_g, lnf_b, hb);
  gemm<float, true, false, false>(stream, hb, tok_emb, d_out, nullptr, nullptr,
                                  NTOK, VOCAB, DMODEL, DMODEL, DMODEL, VOCAB,
                                  1.f);
  (void)in_sizes; (void)n_in; (void)out_size; (void)ws_size;
}
